// GNNTransformerHybrid_30769145708862
// MI455X (gfx1250) — compile-verified, matrix operand order checked by reading, not run on hardware
//
#include <hip/hip_runtime.h>
#include <hip/hip_bf16.h>

// ---------------------------------------------------------------------------
// Types for CDNA5 WMMA (wave32, 16x16x32 bf16 -> f32)
// ---------------------------------------------------------------------------
typedef __bf16 bf16_t;
typedef bf16_t v16bf __attribute__((ext_vector_type(16)));
typedef float  v8f   __attribute__((ext_vector_type(8)));

#define WMMA_BF16(a, b, c) \
    __builtin_amdgcn_wmma_f32_16x16x32_bf16(false, (a), false, (b), (short)0, (c), false, false)

// Load one 16-element bf16 fragment (A or B operand) for lane "hi" group.
// Caller passes p already offset by row*ld + kbase + (hi<<3); elements 0..7 are
// k = kb..kb+7 and elements 8..15 are k = kb+16..kb+23 (contiguous +16).
__device__ __forceinline__ v16bf ldfrag(const __hip_bfloat16* p) {
    union { v16bf v; uint4 u[2]; } f;
    f.u[0] = *reinterpret_cast<const uint4*>(p);
    f.u[1] = *reinterpret_cast<const uint4*>(p + 16);
    return f.v;
}

// ---------------------------------------------------------------------------
// Generic bf16 WMMA GEMM:  C[M,Nc] = A[M,K](bf16) * Wt[Nc,K](bf16, pre-transposed)
//                           + bias, optional ReLU, f32 and/or bf16 outputs.
// Block: 256 threads (8 waves). Block tile 128(M) x 64(N); wave tile 32x32.
// Nc must be a multiple of 64, K a multiple of 32. M is clamped/guarded.
// ---------------------------------------------------------------------------
__global__ __launch_bounds__(256) void k_gemm(
    const __hip_bfloat16* __restrict__ A, const __hip_bfloat16* __restrict__ W,
    const float* __restrict__ bias, float* __restrict__ Cf,
    __hip_bfloat16* __restrict__ Cb, int M, int Nc, int K, int relu)
{
    const int lane = threadIdx.x & 31;
    const int wave = threadIdx.x >> 5;
    const int wm = wave >> 1, wn = wave & 1;
    const int lo = lane & 15, hi = lane >> 4;
    const int row0 = blockIdx.y * 128 + wm * 32;
    const int col0 = blockIdx.x * 64 + wn * 32;

    v8f acc[2][2] = {};

    int rA0 = row0 + lo;        if (rA0 > M - 1) rA0 = M - 1;
    int rA1 = row0 + 16 + lo;   if (rA1 > M - 1) rA1 = M - 1;
    const __hip_bfloat16* pa0 = A + (size_t)rA0 * K + (hi << 3);
    const __hip_bfloat16* pa1 = A + (size_t)rA1 * K + (hi << 3);
    const __hip_bfloat16* pb0 = W + (size_t)(col0 + lo) * K + (hi << 3);
    const __hip_bfloat16* pb1 = W + (size_t)(col0 + 16 + lo) * K + (hi << 3);

    for (int kk = 0; kk < K; kk += 32) {
        v16bf a0 = ldfrag(pa0 + kk);
        v16bf a1 = ldfrag(pa1 + kk);
        v16bf b0 = ldfrag(pb0 + kk);
        v16bf b1 = ldfrag(pb1 + kk);
        acc[0][0] = WMMA_BF16(a0, b0, acc[0][0]);
        acc[0][1] = WMMA_BF16(a0, b1, acc[0][1]);
        acc[1][0] = WMMA_BF16(a1, b0, acc[1][0]);
        acc[1][1] = WMMA_BF16(a1, b1, acc[1][1]);
    }

    #pragma unroll
    for (int ti = 0; ti < 2; ++ti) {
        #pragma unroll
        for (int tj = 0; tj < 2; ++tj) {
            const int n = col0 + tj * 16 + lo;
            const float bv = bias ? bias[n] : 0.f;
            #pragma unroll
            for (int v = 0; v < 8; ++v) {
                const int m = row0 + ti * 16 + v + (hi << 3);
                if (m < M) {
                    float val = acc[ti][tj][v] + bv;
                    if (relu) val = fmaxf(val, 0.f);
                    const size_t off = (size_t)m * Nc + n;
                    if (Cf) Cf[off] = val;
                    if (Cb) Cb[off] = __float2bfloat16(val);
                }
            }
        }
    }
}

// ---------------------------------------------------------------------------
// Fused MHA for S=256, dh=64, all-valid mask. One wave per (b, h, 16-query tile).
// Q,K row-major [N,512] bf16; Vt = V transposed to [B][H][64][256] bf16.
// Out bf16 [N,512].
// ---------------------------------------------------------------------------
__global__ __launch_bounds__(32) void k_attn(
    const __hip_bfloat16* __restrict__ Q, const __hip_bfloat16* __restrict__ Kb,
    const __hip_bfloat16* __restrict__ Vt, __hip_bfloat16* __restrict__ Out)
{
    __shared__ __align__(16) float         Ssc[16 * 256];
    __shared__ __align__(16) __hip_bfloat16 Pl[16 * 256];

    const int lane = threadIdx.x;
    const int qt = blockIdx.x, h = blockIdx.y, b = blockIdx.z;
    const int lo = lane & 15, hi = lane >> 4;
    const int coff = h * 64;

    const size_t qrow = (size_t)(b * 256 + qt * 16 + lo);
    const v16bf a0 = ldfrag(Q + qrow * 512 + coff + (hi << 3));
    const v16bf a1 = ldfrag(Q + qrow * 512 + coff + 32 + (hi << 3));

    // ---- scores = Q K^T / 8, tile by tile, spilled into LDS
    for (int j = 0; j < 16; ++j) {
        const size_t krow = (size_t)(b * 256 + j * 16 + lo);
        v16bf b0 = ldfrag(Kb + krow * 512 + coff + (hi << 3));
        v16bf b1 = ldfrag(Kb + krow * 512 + coff + 32 + (hi << 3));
        v8f c = {};
        c = WMMA_BF16(a0, b0, c);
        c = WMMA_BF16(a1, b1, c);
        #pragma unroll
        for (int v = 0; v < 8; ++v)
            Ssc[(v + (hi << 3)) * 256 + j * 16 + lo] = c[v] * 0.125f;
    }
    __syncthreads();

    // ---- softmax over 256 keys; lane (lo,hi) handles row lo, half hi
    {
        float* Sr = Ssc + lo * 256 + hi * 128;
        float mx = -1e30f;
        for (int i = 0; i < 128; ++i) mx = fmaxf(mx, Sr[i]);
        mx = fmaxf(mx, __shfl_xor(mx, 16));
        float sum = 0.f;
        for (int i = 0; i < 128; ++i) { float e = __expf(Sr[i] - mx); Sr[i] = e; sum += e; }
        sum += __shfl_xor(sum, 16);
        const float inv = 1.f / sum;
        __hip_bfloat16* Pr = Pl + lo * 256 + hi * 128;
        for (int i = 0; i < 128; ++i) Pr[i] = __float2bfloat16(Sr[i] * inv);
    }
    __syncthreads();

    // ---- O = P V    (P: 16x256 from LDS, V^T rows contiguous over keys)
    v8f oacc[4] = {};
    for (int j = 0; j < 8; ++j) {
        v16bf pa = ldfrag(Pl + lo * 256 + j * 32 + (hi << 3));
        #pragma unroll
        for (int t = 0; t < 4; ++t) {
            const size_t vrow = (size_t)((b * 8 + h) * 64 + t * 16 + lo);
            v16bf vb = ldfrag(Vt + vrow * 256 + j * 32 + (hi << 3));
            oacc[t] = WMMA_BF16(pa, vb, oacc[t]);
        }
    }
    #pragma unroll
    for (int t = 0; t < 4; ++t) {
        #pragma unroll
        for (int v = 0; v < 8; ++v) {
            const size_t orow = (size_t)(b * 256 + qt * 16 + v + (hi << 3));
            Out[orow * 512 + h * 64 + t * 16 + lo] = __float2bfloat16(oacc[t][v]);
        }
    }
}

// ---------------------------------------------------------------------------
// Small helper kernels
// ---------------------------------------------------------------------------
__global__ void k_transpose_w(const float* __restrict__ w, __hip_bfloat16* __restrict__ wt,
                              int K, int Ncol, int total) {
    int tid = blockIdx.x * blockDim.x + threadIdx.x;
    if (tid >= total) return;
    int k = tid / Ncol, n = tid - k * Ncol;
    wt[(size_t)n * K + k] = __float2bfloat16(w[tid]);
}

__global__ void k_f32_to_bf16(const float* __restrict__ x, __hip_bfloat16* __restrict__ y, int n) {
    int i = blockIdx.x * blockDim.x + threadIdx.x;
    if (i < n) y[i] = __float2bfloat16(x[i]);
}

__global__ void k_zero(float* p, int n) {
    int i = blockIdx.x * blockDim.x + threadIdx.x;
    if (i < n) p[i] = 0.f;
}

// scatter-add: agg[dst[e]] += h[src[e]]  (4 threads per edge, 128 dims each)
__global__ void k_scatter(const float* __restrict__ h, const int* __restrict__ src,
                          const int* __restrict__ dst, float* __restrict__ agg, int E) {
    int tid = blockIdx.x * blockDim.x + threadIdx.x;
    int e = tid >> 2;
    if (e >= E) return;
    int p = (tid & 3) * 128;
    const float* hs = h + (size_t)src[e] * 512 + p;
    float* ad = agg + (size_t)dst[e] * 512 + p;
    for (int i = 0; i < 128; ++i) atomicAdd(ad + i, hs[i]);
}

// LayerNorm over 512-wide rows; one wave per row. Y = LN(X + Add)*g + b
__global__ __launch_bounds__(32) void k_ln(
    const float* __restrict__ X, const float* __restrict__ Add,
    const float* __restrict__ gam, const float* __restrict__ bet,
    float* __restrict__ Yf, __hip_bfloat16* __restrict__ Yb, int relu)
{
    const int row = blockIdx.x;
    const int lane = threadIdx.x;
    const size_t off = (size_t)row * 512 + lane * 16;
    float v[16];
    float s = 0.f;
    #pragma unroll
    for (int i = 0; i < 16; ++i) {
        float t = X[off + i];
        if (Add) t += Add[off + i];
        v[i] = t; s += t;
    }
    #pragma unroll
    for (int o = 16; o >= 1; o >>= 1) s += __shfl_xor(s, o);
    const float mu = s * (1.f / 512.f);
    float vs = 0.f;
    #pragma unroll
    for (int i = 0; i < 16; ++i) { float d = v[i] - mu; vs += d * d; }
    #pragma unroll
    for (int o = 16; o >= 1; o >>= 1) vs += __shfl_xor(vs, o);
    const float r = rsqrtf(vs * (1.f / 512.f) + 1e-5f);
    #pragma unroll
    for (int i = 0; i < 16; ++i) {
        const int d = lane * 16 + i;
        float y = (v[i] - mu) * r * gam[d] + bet[d];
        if (relu) y = fmaxf(y, 0.f);
        Yf[off + i] = y;
        if (Yb) Yb[off + i] = __float2bfloat16(y);
    }
}

// sincos positional encoding add (+ bf16 mirror)
__global__ void k_pe_add(float* __restrict__ T, __hip_bfloat16* __restrict__ Tb, int total) {
    int tid = blockIdx.x * blockDim.x + threadIdx.x;
    if (tid >= total) return;
    int row = tid >> 9, d = tid & 511;
    int s = row & 255;
    int i = d >> 1;
    float ang = (float)s * __expf((float)(2 * i) * (-9.210340371976184f / 512.f));
    float pe = (d & 1) ? __cosf(ang) : __sinf(ang);
    float v = T[tid] + pe;
    T[tid] = v;
    Tb[tid] = __float2bfloat16(v);
}

// V [N,512] bf16 -> Vt [B][H][64][256] bf16
__global__ void k_transpose_v(const __hip_bfloat16* __restrict__ V,
                              __hip_bfloat16* __restrict__ Vt, int total) {
    int tid = blockIdx.x * blockDim.x + threadIdx.x;
    if (tid >= total) return;
    int node = tid >> 9, col = tid & 511;
    int b = node >> 8, s = node & 255;
    int h = col >> 6, d = col & 63;
    Vt[((size_t)((b * 8 + h) * 64 + d)) * 256 + s] = V[tid];
}

__global__ void k_concat(const __hip_bfloat16* __restrict__ a, const __hip_bfloat16* __restrict__ b,
                         __hip_bfloat16* __restrict__ c, int total) {
    int tid = blockIdx.x * blockDim.x + threadIdx.x;
    if (tid >= total) return;
    int row = tid >> 10, col = tid & 1023;
    c[tid] = (col < 512) ? a[(size_t)row * 512 + col] : b[(size_t)row * 512 + col - 512];
}

// gating: fw = softmax(g1 @ w2 + b2) over 2 outputs, per node
__global__ void k_gate(const float* __restrict__ g1, const float* __restrict__ w2,
                       const float* __restrict__ b2, float* __restrict__ fw, int n) {
    int node = blockIdx.x * blockDim.x + threadIdx.x;
    if (node >= n) return;
    const float* r = g1 + (size_t)node * 512;
    float z0 = b2[0], z1 = b2[1];
    for (int i = 0; i < 512; ++i) { float x = r[i]; z0 += x * w2[2 * i]; z1 += x * w2[2 * i + 1]; }
    float m = fmaxf(z0, z1);
    float e0 = __expf(z0 - m), e1 = __expf(z1 - m);
    float inv = 1.f / (e0 + e1);
    fw[2 * node] = e0 * inv;
    fw[2 * node + 1] = e1 * inv;
}

__global__ void k_fuse(const float* __restrict__ eg, const float* __restrict__ et,
                       const float* __restrict__ fw, float* __restrict__ Ff,
                       __hip_bfloat16* __restrict__ Fb, int total) {
    int tid = blockIdx.x * blockDim.x + threadIdx.x;
    if (tid >= total) return;
    int row = tid >> 9;
    float v = fw[2 * row] * eg[tid] + fw[2 * row + 1] * et[tid];
    Ff[tid] = v;
    Fb[tid] = __float2bfloat16(v);
}

// mean over 256 nodes per graph: q[b,d] = mean_s o[b*256+s, d]
__global__ __launch_bounds__(512) void k_meanpool(const float* __restrict__ o,
                                                  float* __restrict__ qf,
                                                  __hip_bfloat16* __restrict__ qb) {
    int b = blockIdx.x, d = threadIdx.x;
    float s = 0.f;
    for (int i = 0; i < 256; ++i) s += o[(size_t)(b * 256 + i) * 512 + d];
    s *= (1.f / 256.f);
    qf[b * 512 + d] = s;
    qb[b * 512 + d] = __float2bfloat16(s);
}

// pooled attention: Sq=1 query per (b,h) over 256 keys (plain VALU, tiny)
__global__ __launch_bounds__(256) void k_pool_attn(
    const float* __restrict__ qp, const float* __restrict__ kp,
    const float* __restrict__ vp, __hip_bfloat16* __restrict__ ctx_b)
{
    __shared__ float sm[256];
    __shared__ float red[256];
    int h = blockIdx.x, b = blockIdx.y, t = threadIdx.x;
    const float* q = qp + b * 512 + h * 64;
    const float* k = kp + (size_t)(b * 256 + t) * 512 + h * 64;
    float s = 0.f;
    for (int d = 0; d < 64; ++d) s += q[d] * k[d];
    s *= 0.125f;
    red[t] = s;
    __syncthreads();
    for (int o = 128; o > 0; o >>= 1) { if (t < o) red[t] = fmaxf(red[t], red[t + o]); __syncthreads(); }
    float mx = red[0];
    __syncthreads();
    float e = __expf(s - mx);
    sm[t] = e; red[t] = e;
    __syncthreads();
    for (int o = 128; o > 0; o >>= 1) { if (t < o) red[t] += red[t + o]; __syncthreads(); }
    float sum = red[0];
    __syncthreads();
    if (t < 64) {
        float acc = 0.f;
        for (int j = 0; j < 256; ++j) acc += sm[j] * vp[(size_t)(b * 256 + j) * 512 + h * 64 + t];
        ctx_b[b * 512 + h * 64 + t] = __float2bfloat16(acc / sum);
    }
}

// ---------------------------------------------------------------------------
// Host orchestration
// ---------------------------------------------------------------------------
extern "C" void kernel_launch(void* const* d_in, const int* in_sizes, int n_in,
                              void* d_out, int out_size, void* d_ws, size_t ws_size,
                              hipStream_t stream) {
    (void)n_in; (void)out_size; (void)ws_size;
    const int NB = 16, DH = 512, NH = 8;
    const int NN = 4096;             // total nodes
    const int E = in_sizes[1] / 2;   // edges

    // Input leaf order: JAX pytree (sorted dict keys, DFS).
    // 0:batch 1:edge_index 2..19:fusion 20..29:gnn 30..33:out 34..41:pool
    // 42..141:tr 142:x
    auto FP = [&](int i) { return (const float*)d_in[i]; };
    const int* e_src = (const int*)d_in[1];
    const int* e_dst = e_src + E;
    const float* x_f32 = FP(142);

    // ---- bump allocator over workspace
    char* wsp = (char*)d_ws;
    auto alloc = [&](size_t bytes) -> void* {
        void* p = (void*)wsp;
        wsp += (bytes + 255) & ~(size_t)255;
        return p;
    };

    // ---- launch helpers
    auto gemm = [&](const __hip_bfloat16* A, const __hip_bfloat16* W, const float* bias,
                    float* Cf, __hip_bfloat16* Cb, int M, int Nc, int K, int relu) {
        dim3 g(Nc / 64, (M + 127) / 128);
        k_gemm<<<g, 256, 0, stream>>>(A, W, bias, Cf, Cb, M, Nc, K, relu);
    };
    auto XW = [&](int idx, int K, int Ncol) -> const __hip_bfloat16* {
        __hip_bfloat16* dst = (__hip_bfloat16*)alloc((size_t)K * Ncol * 2);
        int total = K * Ncol;
        k_transpose_w<<<(total + 255) / 256, 256, 0, stream>>>(FP(idx), dst, K, Ncol, total);
        return dst;
    };

    // ---- pre-transpose all weights to bf16 [N][K]
    const __hip_bfloat16* WT_fk = XW(3, 512, 512);
    const __hip_bfloat16* WT_fo = XW(5, 512, 512);
    const __hip_bfloat16* WT_fq = XW(7, 512, 512);
    const __hip_bfloat16* WT_fv = XW(9, 512, 512);
    const __hip_bfloat16* WT_fe1 = XW(11, 512, 512);
    const __hip_bfloat16* WT_fe2 = XW(13, 512, 512);
    const __hip_bfloat16* WT_wg1 = XW(17, 1024, 512);
    const __hip_bfloat16* WT_gin = XW(21, 256, 512);
    const __hip_bfloat16* WT_gout = XW(29, 512, 512);
    const __hip_bfloat16* WT_l1 = XW(31, 512, 512);
    const __hip_bfloat16* WT_l2 = XW(33, 512, 512);
    const __hip_bfloat16* WT_pk = XW(35, 512, 512);
    const __hip_bfloat16* WT_po = XW(37, 512, 512);
    const __hip_bfloat16* WT_pq = XW(39, 512, 512);
    const __hip_bfloat16* WT_pv = XW(41, 512, 512);
    const __hip_bfloat16* WT_tin = XW(43, 256, 512);
    const __hip_bfloat16* WT_ak[6], *WT_ao[6], *WT_aq[6], *WT_av[6], *WT_ff1[6], *WT_ff2[6];
    for (int L = 0; L < 6; ++L) {
        int base = 44 + 16 * L;
        WT_ak[L] = XW(base + 1, 512, 512);
        WT_ao[L] = XW(base + 3, 512, 512);
        WT_aq[L] = XW(base + 5, 512, 512);
        WT_av[L] = XW(base + 7, 512, 512);
        WT_ff1[L] = XW(base + 9, 512, 2048);
        WT_ff2[L] = XW(base + 11, 2048, 512);
    }
    const __hip_bfloat16* WT_tout = XW(141, 512, 512);

    // ---- activation buffers
    const size_t R512f = (size_t)NN * 512 * 4, R512b = (size_t)NN * 512 * 2;
    __hip_bfloat16* xb = (__hip_bfloat16*)alloc((size_t)NN * 256 * 2);
    float* F_h = (float*)alloc(R512f);
    float* F_tmp = (float*)alloc(R512f);           // agg / attn-out / ff-out / enh / wg1-out
    __hip_bfloat16* B_h = (__hip_bfloat16*)alloc(R512b);
    float* F_t = (float*)alloc(R512f);
    __hip_bfloat16* B_t = (__hip_bfloat16*)alloc(R512b);
    __hip_bfloat16* B_q = (__hip_bfloat16*)alloc(R512b);
    __hip_bfloat16* B_k = (__hip_bfloat16*)alloc(R512b);
    __hip_bfloat16* B_v = (__hip_bfloat16*)alloc(R512b);
    __hip_bfloat16* B_vt = (__hip_bfloat16*)alloc(R512b);
    __hip_bfloat16* B_attn = (__hip_bfloat16*)alloc(R512b);
    __hip_bfloat16* B_h1 = (__hip_bfloat16*)alloc((size_t)NN * 2048 * 2);
    float* F_gnn = (float*)alloc(R512f);
    __hip_bfloat16* B_gnn = (__hip_bfloat16*)alloc(R512b);
    float* F_tr = (float*)alloc(R512f);
    __hip_bfloat16* B_tr = (__hip_bfloat16*)alloc(R512b);
    __hip_bfloat16* B_comb = (__hip_bfloat16*)alloc((size_t)NN * 1024 * 2);
    float* FW = (float*)alloc((size_t)NN * 2 * 4);
    float* F_eg = (float*)alloc(R512f);
    float* F_et = (float*)alloc(R512f);
    float* F_fused = (float*)alloc(R512f);
    __hip_bfloat16* B_fused = (__hip_bfloat16*)alloc(R512b);
    __hip_bfloat16* B_e1 = (__hip_bfloat16*)alloc(R512b);
    __hip_bfloat16* B_o1 = (__hip_bfloat16*)alloc(R512b);
    float* F_o = (float*)alloc(R512f);
    __hip_bfloat16* B_o = (__hip_bfloat16*)alloc(R512b);
    float* F_qm = (float*)alloc((size_t)NB * 512 * 4);
    __hip_bfloat16* B_qm = (__hip_bfloat16*)alloc((size_t)NB * 512 * 2);
    float* F_qp = (float*)alloc((size_t)NB * 512 * 4);
    float* F_kp = F_eg;   // reuse (dead after fusion mix)
    float* F_vp = F_et;   // reuse
    __hip_bfloat16* B_ctx = (__hip_bfloat16*)alloc((size_t)NB * 512 * 2);

    const dim3 attn_grid(16, NH, NB);
    const int tot512 = NN * 512;

    // ---- input conversion
    k_f32_to_bf16<<<(NN * 256 + 255) / 256, 256, 0, stream>>>(x_f32, xb, NN * 256);

    // ================= GNN branch =================
    gemm(xb, WT_gin, FP(20), F_h, nullptr, NN, 512, 256, 0);
    for (int l = 0; l < 3; ++l) {
        k_zero<<<(tot512 + 255) / 256, 256, 0, stream>>>(F_tmp, tot512);
        k_scatter<<<(E * 4 + 255) / 256, 256, 0, stream>>>(F_h, e_src, e_dst, F_tmp, E);
        // h = relu(LN(agg + h))
        k_ln<<<NN, 32, 0, stream>>>(F_tmp, F_h, FP(23 + 2 * l), FP(22 + 2 * l), F_h, B_h, 1);
    }
    gemm(B_h, WT_gout, FP(28), F_gnn, B_gnn, NN, 512, 512, 0);

    // ================= Transformer branch =================
    gemm(xb, WT_tin, FP(42), F_t, nullptr, NN, 512, 256, 0);
    k_pe_add<<<(tot512 + 255) / 256, 256, 0, stream>>>(F_t, B_t, tot512);
    for (int L = 0; L < 6; ++L) {
        int base = 44 + 16 * L;
        gemm(B_t, WT_aq[L], FP(base + 4), nullptr, B_q, NN, 512, 512, 0);
        gemm(B_t, WT_ak[L], FP(base + 0), nullptr, B_k, NN, 512, 512, 0);
        gemm(B_t, WT_av[L], FP(base + 6), nullptr, B_v, NN, 512, 512, 0);
        k_transpose_v<<<(tot512 + 255) / 256, 256, 0, stream>>>(B_v, B_vt, tot512);
        k_attn<<<attn_grid, 32, 0, stream>>>(B_q, B_k, B_vt, B_attn);
        gemm(B_attn, WT_ao[L], FP(base + 2), F_tmp, nullptr, NN, 512, 512, 0);
        k_ln<<<NN, 32, 0, stream>>>(F_tmp, F_t, FP(base + 13), FP(base + 12), F_t, B_t, 0);
        gemm(B_t, WT_ff1[L], FP(base + 8), nullptr, B_h1, NN, 2048, 512, 1);
        gemm(B_h1, WT_ff2[L], FP(base + 10), F_tmp, nullptr, NN, 512, 2048, 0);
        k_ln<<<NN, 32, 0, stream>>>(F_tmp, F_t, FP(base + 15), FP(base + 14), F_t, B_t, 0);
    }
    gemm(B_t, WT_tout, FP(140), F_tr, B_tr, NN, 512, 512, 0);

    // ================= Fusion =================
    k_concat<<<(NN * 1024 + 255) / 256, 256, 0, stream>>>(B_gnn, B_tr, B_comb, NN * 1024);
    gemm(B_comb, WT_wg1, FP(16), F_tmp, nullptr, NN, 512, 1024, 1);
    k_gate<<<(NN + 255) / 256, 256, 0, stream>>>(F_tmp, FP(19), FP(18), FW, NN);
    // eg = attn(Q=gnn, K/V=tr)
    gemm(B_gnn, WT_fq, FP(6), nullptr, B_q, NN, 512, 512, 0);
    gemm(B_tr, WT_fk, FP(2), nullptr, B_k, NN, 512, 512, 0);
    gemm(B_tr, WT_fv, FP(8), nullptr, B_v, NN, 512, 512, 0);
    k_transpose_v<<<(tot512 + 255) / 256, 256, 0, stream>>>(B_v, B_vt, tot512);
    k_attn<<<attn_grid, 32, 0, stream>>>(B_q, B_k, B_vt, B_attn);
    gemm(B_attn, WT_fo, FP(4), F_eg, nullptr, NN, 512, 512, 0);
    // et = attn(Q=tr, K/V=gnn)
    gemm(B_tr, WT_fq, FP(6), nullptr, B_q, NN, 512, 512, 0);
    gemm(B_gnn, WT_fk, FP(2), nullptr, B_k, NN, 512, 512, 0);
    gemm(B_gnn, WT_fv, FP(8), nullptr, B_v, NN, 512, 512, 0);
    k_transpose_v<<<(tot512 + 255) / 256, 256, 0, stream>>>(B_v, B_vt, tot512);
    k_attn<<<attn_grid, 32, 0, stream>>>(B_q, B_k, B_vt, B_attn);
    gemm(B_attn, WT_fo, FP(4), F_et, nullptr, NN, 512, 512, 0);
    // fused = fw0*eg + fw1*et ; enh = fe2(relu(fe1(fused))) ; fused = LN(fused+enh)
    k_fuse<<<(tot512 + 255) / 256, 256, 0, stream>>>(F_eg, F_et, FW, F_fused, B_fused, tot512);
    gemm(B_fused, WT_fe1, FP(10), nullptr, B_e1, NN, 512, 512, 1);
    gemm(B_e1, WT_fe2, FP(12), F_tmp, nullptr, NN, 512, 512, 0);
    k_ln<<<NN, 32, 0, stream>>>(F_tmp, F_fused, FP(15), FP(14), F_fused, B_fused, 0);

    // ================= Output MLP + attention pooling =================
    gemm(B_fused, WT_l1, FP(30), nullptr, B_o1, NN, 512, 512, 1);
    gemm(B_o1, WT_l2, FP(32), F_o, B_o, NN, 512, 512, 0);
    k_meanpool<<<NB, 512, 0, stream>>>(F_o, F_qm, B_qm);
    gemm(B_qm, WT_pq, FP(38), F_qp, nullptr, NB, 512, 512, 0);
    gemm(B_o, WT_pk, FP(34), F_kp, nullptr, NN, 512, 512, 0);
    gemm(B_o, WT_pv, FP(40), F_vp, nullptr, NN, 512, 512, 0);
    k_pool_attn<<<dim3(NH, NB), 256, 0, stream>>>(F_qp, F_kp, F_vp, B_ctx);
    gemm(B_ctx, WT_po, FP(36), (float*)d_out, nullptr, NB, 512, 512, 0);
}
